// SingleLayerGLIFRSNN_1941325217856
// MI455X (gfx1250) — matrix operand order repaired and checked by reading, not compiled, not run on hardware
//
#include <hip/hip_runtime.h>

typedef __attribute__((ext_vector_type(16))) _Float16 v16h;
typedef __attribute__((ext_vector_type(8)))  _Float16 v8h;
typedef __attribute__((ext_vector_type(8)))  float    v8f;

namespace {
constexpr int T_STEPS  = 1000;
constexpr int BATCH    = 64;
constexpr int DIN      = 128;
constexpr int NN       = 512;
constexpr int NOUT     = 10;
constexpr int SSTR     = 520;     // padded row stride in f16 units (1040B == 4 banks mod 64)
constexpr int LDSROWS  = 288;     // W_rec rows pinned in LDS (waves 0..8)
constexpr int T_START  = 200;     // int(T * (1 - RESPONSE_WINDOW))

constexpr float V_TH      = -45.0f;
constexpr float V_RESET   = -60.0f;
constexpr float A_MEM     = 0.05f;                 // DT/TAU_MEM
constexpr float E_SYN     = 0.818730753077982f;    // exp(-DT/TAU_SYN)
constexpr float DTS       = 0.2f;                  // DT/TAU_SYN
constexpr float E_ASC     = 0.998572448628563f;    // exp(-DT/TAU_ADAPT)
constexpr float ASC_AMPC  = -0.2f;
constexpr int   REF_STEPS = 5;
constexpr float ALPHA     = 0.951229424500714f;    // exp(-DT/READOUT_TAU)
}

__device__ __forceinline__ v16h make16(v8h lo, v8h hi) {
  v16h r;
#pragma unroll
  for (int i = 0; i < 8; ++i) { r[i] = lo[i]; r[i + 8] = hi[i]; }
  return r;
}

__device__ __forceinline__ v8f wmma_f16(v16h a, v16h b, v8f c) {
  return __builtin_amdgcn_wmma_f32_16x16x32_f16(false, a, false, b,
                                                (short)0, c, false, false);
}

// ---------------------------------------------------------------------------
// Kernel 1: convert W_rec (N x N) and W_in (N x D) to f16 in workspace.
// Row-major is kept: row n of W_rec/W_in is exactly column n of the WMMA B
// matrix (B[k][n] = W[n][k]), giving contiguous per-lane b128 loads.
// ---------------------------------------------------------------------------
__global__ void snn_prep_kernel(const float* __restrict__ Wrec,
                                const float* __restrict__ Win,
                                _Float16* __restrict__ wrec_h,
                                _Float16* __restrict__ win_h) {
  const int i = blockIdx.x * blockDim.x + threadIdx.x;
  if (i < NN * NN) wrec_h[i] = (_Float16)Wrec[i];
  if (i < NN * DIN) win_h[i] = (_Float16)Win[i];
}

// ---------------------------------------------------------------------------
// Kernel 2: chip-wide input-drive GEMM (no sequential dependency):
//   I_ext = (x * input_scale) @ W_in.T, shape (T*B, N), K = D = 128.
// Result is staged in the d_out "voltage" region (same (t,b,n) addresses are
// overwritten with voltage by the scan, by the very lane that consumed them).
// grid = (4000 M-tiles, 4 N-groups) x 256 threads (8 waves, 1 tile each).
// ---------------------------------------------------------------------------
__global__ void snn_iext_kernel(const float* __restrict__ x,
                                const _Float16* __restrict__ win_h,
                                const float* __restrict__ in_scale_p,
                                float* __restrict__ iext) {
  const int lane  = threadIdx.x & 31;
  const int wv    = threadIdx.x >> 5;            // 0..7
  const int half  = lane >> 4;
  const int nlo   = lane & 15;
  const int mtile = blockIdx.x;                  // 0..3999
  const int ntile = blockIdx.y * 8 + wv;         // 0..31
  const int nc    = ntile * 16 + nlo;

  const float sc = *in_scale_p;
  const v8h* __restrict__ wrow = (const v8h*)(win_h + (size_t)nc * DIN);
  const float* __restrict__ xr = x + ((size_t)mtile * 16 + nlo) * DIN;

  v8f acc = {};
#pragma unroll
  for (int ks = 0; ks < 4; ++ks) {
    const int k0 = ks * 32;
    const float4 x0 = *(const float4*)(xr + k0 + 8 * half);
    const float4 x1 = *(const float4*)(xr + k0 + 8 * half + 4);
    const float4 x2 = *(const float4*)(xr + k0 + 16 + 8 * half);
    const float4 x3 = *(const float4*)(xr + k0 + 16 + 8 * half + 4);
    v16h a;
    a[0]  = (_Float16)x0.x; a[1]  = (_Float16)x0.y; a[2]  = (_Float16)x0.z; a[3]  = (_Float16)x0.w;
    a[4]  = (_Float16)x1.x; a[5]  = (_Float16)x1.y; a[6]  = (_Float16)x1.z; a[7]  = (_Float16)x1.w;
    a[8]  = (_Float16)x2.x; a[9]  = (_Float16)x2.y; a[10] = (_Float16)x2.z; a[11] = (_Float16)x2.w;
    a[12] = (_Float16)x3.x; a[13] = (_Float16)x3.y; a[14] = (_Float16)x3.z; a[15] = (_Float16)x3.w;
    const int kb = 4 * ks + half;
    const v16h b = make16(wrow[kb], wrow[kb + 2]);
    acc = wmma_f16(a, b, acc);
  }
#pragma unroll
  for (int r = 0; r < 8; ++r) {
    const size_t row = (size_t)mtile * 16 + r + 8 * half;   // = t*BATCH + b
    iext[row * NN + nc] = acc[r] * sc;
  }
}

// ---------------------------------------------------------------------------
// Kernel 3: persistent scan. grid = 4 blocks (one 16-row batch block each),
// block = 512 threads = 16 wave32 waves; wave w owns output columns
// [32w, 32w+32) (two 16x16 WMMA tiles) for all 16 batch rows.
// CDNA5 LDS (320KB/WGP) pins 288 of 512 W_rec rows -> waves 0..8 feed their
// B matrices from ds_load_b128; input drive comes from precomputed I_ext
// (read from vlt_iext, then overwritten in place with voltage).
// ---------------------------------------------------------------------------
__launch_bounds__(512, 1)
__global__ void snn_scan_kernel(const _Float16* __restrict__ wrec_h,
                                float* __restrict__ spk,
                                float* vlt_iext,          // NOT restrict: read I_ext, write voltage
                                float* __restrict__ ph,
                                float* __restrict__ fl,
                                float* __restrict__ avg_out) {
  __shared__ __align__(16) _Float16 sbuf[16 * SSTR];        //  16.25 KB spikes
  __shared__ __align__(16) _Float16 wslc[LDSROWS * SSTR];   // 292.5 KB W_rec rows

  const int tid  = threadIdx.x;
  const int lane = tid & 31;        // wave32
  const int wv   = tid >> 5;        // 0..15
  const int half = lane >> 4;       // K-half selector (A/B layouts)
  const int nlo  = lane & 15;       // A row (M) for loads; C/D column (N)
  const int bb   = blockIdx.x;      // batch block: rows [16*bb, 16*bb+16)
  const int n0   = wv * 32;         // first output column of this wave
  const int nc0  = n0 + nlo;        // tile 0 column
  const int nc1  = n0 + 16 + nlo;   // tile 1 column

  // zero spike staging (s_prev = 0 at t = 0)
  for (int i = tid; i < 16 * SSTR; i += 512) sbuf[i] = (_Float16)0.0f;
  // pin first LDSROWS rows of W_rec f16 into LDS (padded stride)
  for (int i = tid; i < LDSROWS * (NN / 8); i += 512) {
    const int row = i >> 6;           // / (NN/8)
    const int seg = i & 63;           // v8h segment within row
    ((v8h*)(wslc + row * SSTR))[seg] = ((const v8h*)(wrec_h + (size_t)row * NN))[seg];
  }
  __syncthreads();

  // Per-lane persistent state: [tile][C-vgpr r] -> (batch row = r+8*half, col = ncX)
  float v_[2][8], asc_[2][8], rr_[2][8], psc_[2][8], f_[2][8], avg_[2][8];
  int   ref_[2][8];
#pragma unroll
  for (int tl = 0; tl < 2; ++tl)
#pragma unroll
    for (int r = 0; r < 8; ++r) {
      v_[tl][r] = V_RESET; asc_[tl][r] = 0.0f; rr_[tl][r] = 0.0f;
      psc_[tl][r] = 0.0f;  f_[tl][r] = 0.0f;   avg_[tl][r] = 0.0f;
      ref_[tl][r] = 0;
    }

  const bool b_in_lds = (n0 + 32) <= LDSROWS;   // wave-uniform B source select
  const v8h* __restrict__ brow0 = (const v8h*)(wrec_h + (size_t)nc0 * NN);
  const v8h* __restrict__ brow1 = (const v8h*)(wrec_h + (size_t)nc1 * NN);
  const v8h* __restrict__ lrow0 = (const v8h*)(wslc + nc0 * SSTR);
  const v8h* __restrict__ lrow1 = (const v8h*)(wslc + nc1 * SSTR);

  if (!b_in_lds) {  // warm the L2-streamed W_rec rows (reused 1000x)
    __builtin_prefetch((const void*)brow0, 0, 3);
    __builtin_prefetch((const void*)((const char*)brow0 + 512), 0, 3);
    __builtin_prefetch((const void*)brow1, 0, 3);
    __builtin_prefetch((const void*)((const char*)brow1 + 512), 0, 3);
  }

#pragma unroll 1
  for (int t = 0; t < T_STEPS; ++t) {
    v8f accr0 = {}, accr1 = {};   // recurrent GEMM accumulators

    // ---- recurrent GEMM: rec = s_prev @ W_rec.T, K = 512 ----
    if (b_in_lds) {
#pragma unroll 4
      for (int ks = 0; ks < 16; ++ks) {
        const int k0 = ks * 32;
        const v8h alo = *(const v8h*)(sbuf + nlo * SSTR + k0 + 16 * half);
        const v8h ahi = *(const v8h*)(sbuf + nlo * SSTR + k0 + 16 * half + 8);
        const v16h a = make16(alo, ahi);
        const int kb = 4 * ks + 2 * half;
        const v16h b0 = make16(lrow0[kb], lrow0[kb + 1]);
        const v16h b1 = make16(lrow1[kb], lrow1[kb + 1]);
        accr0 = wmma_f16(a, b0, accr0);
        accr1 = wmma_f16(a, b1, accr1);
      }
    } else {
#pragma unroll 4
      for (int ks = 0; ks < 16; ++ks) {
        const int k0 = ks * 32;
        const v8h alo = *(const v8h*)(sbuf + nlo * SSTR + k0 + 16 * half);
        const v8h ahi = *(const v8h*)(sbuf + nlo * SSTR + k0 + 16 * half + 8);
        const v16h a = make16(alo, ahi);
        const int kb = 4 * ks + 2 * half;
        const v16h b0 = make16(brow0[kb], brow0[kb + 1]);
        const v16h b1 = make16(brow1[kb], brow1[kb + 1]);
        accr0 = wmma_f16(a, b0, accr0);
        accr1 = wmma_f16(a, b1, accr1);
      }
    }

    // ---- load precomputed input drive I_ext(t, b, n) (scaled already) ----
    float ai0[8], ai1[8];
#pragma unroll
    for (int r = 0; r < 8; ++r) {
      const size_t base = ((size_t)t * BATCH + (bb * 16 + r + 8 * half)) * NN;
      ai0[r] = vlt_iext[base + nc0];
      ai1[r] = vlt_iext[base + nc1];
    }

    __syncthreads();   // all waves done reading sbuf (s_prev)

    // ---- GLIF3 state update + filter + outputs ----
#pragma unroll
    for (int tl = 0; tl < 2; ++tl) {
      const int nc = n0 + tl * 16 + nlo;
      const v8f ar = tl ? accr1 : accr0;
#pragma unroll
      for (int r = 0; r < 8; ++r) {
        const int mrow = r + 8 * half;          // batch row within block
        const float rec = ar[r];
        const float it  = tl ? ai1[r] : ai0[r];

        rr_[tl][r]  = rr_[tl][r] * E_SYN + rec;
        psc_[tl][r] = psc_[tl][r] * E_SYN + DTS * rr_[tl][r];
        const float I = it + psc_[tl][r] + asc_[tl][r];
        float vv = v_[tl][r] + A_MEM * (V_RESET - v_[tl][r]) + A_MEM * I;
        const bool inref = ref_[tl][r] > 0;
        if (inref) vv = V_RESET;
        const float s = (!inref && (vv - V_TH >= 0.0f)) ? 1.0f : 0.0f;
        if (s > 0.0f) { vv = V_RESET; ref_[tl][r] = REF_STEPS; }
        else          { ref_[tl][r] = inref ? (ref_[tl][r] - 1) : 0; }
        v_[tl][r]   = vv;
        asc_[tl][r] = asc_[tl][r] * E_ASC + ASC_AMPC * s;
        const float ff = (t == 0) ? s : (ALPHA * f_[tl][r] + (1.0f - ALPHA) * s);
        f_[tl][r] = ff;
        if (t >= T_START) avg_[tl][r] += ff;

        const size_t gi = ((size_t)t * BATCH + (bb * 16 + mrow)) * NN + nc;
        spk[gi]      = s;
        vlt_iext[gi] = vv;     // overwrites the I_ext value this lane just consumed
        ph[gi]       = psc_[tl][r];
        fl[gi]       = ff;

        sbuf[mrow * SSTR + nc] = (_Float16)s;   // stage spike for next step's A
      }
    }

    __syncthreads();   // spikes visible to all waves
  }

  // ---- response-window mean -> workspace ----
#pragma unroll
  for (int tl = 0; tl < 2; ++tl)
#pragma unroll
    for (int r = 0; r < 8; ++r) {
      const int nc = n0 + tl * 16 + nlo;
      avg_out[(size_t)(bb * 16 + r + 8 * half) * NN + nc] =
          avg_[tl][r] * (1.0f / (float)(T_STEPS - T_START));
    }
}

// ---------------------------------------------------------------------------
// Kernel 4: readout  output = (avg * output_scale) @ W_out.T + b_out  (64x10)
// ---------------------------------------------------------------------------
__global__ void snn_out_kernel(const float* __restrict__ avg,
                               const float* __restrict__ Wout,
                               const float* __restrict__ bout,
                               const float* __restrict__ out_scale_p,
                               float* __restrict__ out) {
  const int tid = blockIdx.x * blockDim.x + threadIdx.x;
  if (tid >= BATCH * NOUT) return;
  const int b = tid / NOUT, o = tid % NOUT;
  const float sc = *out_scale_p;
  float acc = 0.0f;
  for (int n = 0; n < NN; ++n) acc += avg[b * NN + n] * Wout[o * NN + n];
  out[tid] = acc * sc + bout[o];
}

extern "C" void kernel_launch(void* const* d_in, const int* in_sizes, int n_in,
                              void* d_out, int out_size, void* d_ws, size_t ws_size,
                              hipStream_t stream) {
  (void)in_sizes; (void)n_in; (void)out_size; (void)ws_size;
  const float* x         = (const float*)d_in[0];  // (T,B,D)
  const float* W_in      = (const float*)d_in[1];  // (N,D)
  const float* in_scale  = (const float*)d_in[2];  // scalar
  const float* W_rec     = (const float*)d_in[3];  // (N,N)
  const float* W_out     = (const float*)d_in[4];  // (O,N)
  const float* b_out     = (const float*)d_in[5];  // (O,)
  const float* out_scale = (const float*)d_in[6];  // scalar

  // d_out: output(640) | spikes | voltage | psc | filtered (each T*B*N f32)
  float* out = (float*)d_out;
  const size_t TBN = (size_t)T_STEPS * BATCH * NN;   // 32,768,000
  float* spk = out + BATCH * NOUT;
  float* vlt = spk + TBN;      // doubles as I_ext staging
  float* ph  = vlt + TBN;
  float* fl  = ph + TBN;

  // workspace: W_rec f16 (512KB) | W_in f16 (128KB) | avg f32 (128KB)
  char* ws = (char*)d_ws;
  _Float16* wrec_h = (_Float16*)ws;
  _Float16* win_h  = (_Float16*)(ws + (size_t)NN * NN * sizeof(_Float16));
  float*    avg    = (float*)(ws + (size_t)NN * NN * sizeof(_Float16)
                                 + (size_t)NN * DIN * sizeof(_Float16));

  snn_prep_kernel<<<(NN * NN + 255) / 256, 256, 0, stream>>>(W_rec, W_in, wrec_h, win_h);
  snn_iext_kernel<<<dim3(T_STEPS * BATCH / 16, 4), 256, 0, stream>>>(x, win_h, in_scale, vlt);
  snn_scan_kernel<<<BATCH / 16, 512, 0, stream>>>(wrec_h, spk, vlt, ph, fl, avg);
  snn_out_kernel<<<(BATCH * NOUT + 127) / 128, 128, 0, stream>>>(avg, W_out, b_out,
                                                                 out_scale, out);
}
